// FeaturePyramidCorrelation_56255481643677
// MI455X (gfx1250) — matrix operand order repaired and checked by reading, not compiled, not run on hardware
//
#include <hip/hip_runtime.h>

typedef __attribute__((ext_vector_type(16))) __bf16 v16bf;
typedef __attribute__((ext_vector_type(2)))  __bf16 v2bf;
typedef __attribute__((ext_vector_type(2)))  float  vf2;
typedef __attribute__((ext_vector_type(8)))  float  v8f;

#define EPSBN 1e-5f

// f32 pair -> packed bf16x2 dword; lowers to v_cvt_pk_bf16_f32
__device__ __forceinline__ v2bf cvt2(float a, float b) {
  vf2 t; t.x = a; t.y = b;
  return __builtin_convertvector(t, v2bf);
}
__device__ __forceinline__ unsigned cvt2u(float a, float b) {
  return __builtin_bit_cast(unsigned, cvt2(a, b));
}

union BfFrag { v2bf h[8]; v16bf v; uint4 q[2]; };

template <int HW>
__launch_bounds__(128)
__global__ void gate_merge_kernel(
    const float* __restrict__ fp, const float* __restrict__ mean,
    const float* __restrict__ w1, const float* __restrict__ b1,
    const float* __restrict__ g1, const float* __restrict__ be1,
    const float* __restrict__ m1, const float* __restrict__ v1,
    const float* __restrict__ w2, const float* __restrict__ b2,
    const float* __restrict__ g2, const float* __restrict__ be2,
    const float* __restrict__ m2, const float* __restrict__ v2,
    float* __restrict__ out)
{
  constexpr int tilesPerBatch = HW / 16;

  // bf16 image of w1: 64 rows x 512 cols -> 64 x 256 packed dwords = 64 KB.
  // After the gate GEMM it is dead; its first 64 floats are reused (behind a
  // barrier) as the per-wave gate buffer so static LDS stays at exactly 64 KB.
  __shared__ unsigned w1lds[64 * 256];

  const int tid  = threadIdx.x;
  const int lane = tid & 31;
  const int wv   = tid >> 5;
  const int h    = lane >> 4;      // lane half (0/1)
  const int m    = lane & 15;      // pixel within tile (A rows / D cols role)

  // ---- stage w1 as bf16 in LDS (once per block, coalesced b128 reads) ----
  for (int i = tid * 2; i < 64 * 256; i += 256) {      // 64 iters/thread
    const float4 q = *(const float4*)(w1 + i * 2);     // 2 dword-pairs
    uint2 d;
    d.x = cvt2u(q.x, q.y);
    d.y = cvt2u(q.z, q.w);
    *(uint2*)&w1lds[i] = d;                            // ds_store_b64
  }
  __syncthreads();

  const int task = blockIdx.x * 4 + wv;          // (batch, pixel-tile) task
  const int b    = task / tilesPerBatch;         // const divisor -> mul/shift
  const int tile = task - b * tilesPerBatch;
  const int pix0 = tile * 16;

  const long fpBase = (long)b * 256 * HW + pix0 + m;

  // D accumulators: 4 N-tiles of 16 out-channels; M = 16 pixels
  v8f acc[4];
  #pragma unroll
  for (int nt = 0; nt < 4; ++nt)
    #pragma unroll
    for (int i = 0; i < 8; ++i) acc[nt][i] = 0.0f;

  // ---- gate GEMM: K = 256 (fp) + 256 (mean), 32 per WMMA -> 64 WMMAs ----
  for (int srcSel = 0; srcSel < 2; ++srcSel) {
    const float* src = srcSel ? (mean + pix0 + m) : (fp + fpBase);
    const int colBase = srcSel * 256;
    for (int kc = 0; kc < 8; ++kc) {
      // A fragment (16x32 bf16): this lane's pixel, channels per ISA layout:
      // VGPR v<4: K = 2v,2v+1 (+8*half); v>=4: K = 16+2(v-4) (+8*half)
      // HW is constexpr -> one base address + 24-bit immediate offsets.
      const float* srck = src + (long)(kc * 32 + 8 * h) * HW;
      BfFrag A;
      #pragma unroll
      for (int v = 0; v < 8; ++v) {
        constexpr long lHW = HW;
        const int d = 2 * v + ((v >= 4) ? 8 : 0);
        A.h[v] = cvt2(srck[d * lHW], srck[(d + 1) * lHW]);
      }
      const int kcol = (colBase + kc * 32 + 8 * h) >> 1;  // packed-dword col
      #pragma unroll
      for (int nt = 0; nt < 4; ++nt) {
        // B fragment (32x16 bf16): col n = nt*16+m (w1 row); lane's 16 bf16
        // are two runs of 8 (K = 8h.., 16+8h..) -> two ds_load_b128
        const unsigned* bp = w1lds + (nt * 16 + m) * 256 + kcol;
        BfFrag Bf;
        Bf.q[0] = *(const uint4*)(bp);
        Bf.q[1] = *(const uint4*)(bp + 8);
        acc[nt] = __builtin_amdgcn_wmma_f32_16x16x32_bf16(
            false, A.v, false, Bf.v, (short)0, acc[nt], false, false);
      }
    }
  }

  // ---- BN1 affine (+conv bias) + ReLU + 64->2 layer partials ----
  float w2a[4], w2b[4], s1v[4], t1v[4];
  #pragma unroll
  for (int nt = 0; nt < 4; ++nt) {
    const int n = nt * 16 + m;                 // this lane's out-channel
    const float s = g1[n] * rsqrtf(v1[n] + EPSBN);
    s1v[nt] = s;
    t1v[nt] = b1[n] * s + be1[n] - m1[n] * s;  // conv bias folded through BN
    w2a[nt] = w2[n];
    w2b[nt] = w2[64 + n];
  }
  const float s2a = g2[0] * rsqrtf(v2[0] + EPSBN);
  const float s2b = g2[1] * rsqrtf(v2[1] + EPSBN);
  const float t2a = b2[0] * s2a + be2[0] - m2[0] * s2a;
  const float t2b = b2[1] * s2b + be2[1] - m2[1] * s2b;

  float myGv = 0.0f;                          // gate value for pixel 8h+m
  #pragma unroll
  for (int r = 0; r < 8; ++r) {               // pixel = r + 8*h (D layout)
    float a0 = 0.f, a1 = 0.f;
    #pragma unroll
    for (int nt = 0; nt < 4; ++nt) {
      const float hp = fmaxf(acc[nt][r] * s1v[nt] + t1v[nt], 0.f);
      a0 += hp * w2a[nt];
      a1 += hp * w2b[nt];
    }
    // reduce the 64 channels across the 16 lanes of this half
    #pragma unroll
    for (int off = 1; off < 16; off <<= 1) {   // xor masks < 16: stay in half
      a0 += __shfl_xor(a0, off);
      a1 += __shfl_xor(a1, off);
    }
    const float L0 = a0 * s2a + t2a;
    const float L1 = a1 * s2b + t2b;
    const float gv = 1.0f / (1.0f + __expf(L1 - L0));  // softmax_0 of [L0,L1]
    if (m == r) myGv = gv;
  }

  __syncthreads();                 // all waves done reading w1lds
  float* gbuf = (float*)w1lds;     // reuse first 256 B of dead weight image
  if (m < 8) gbuf[wv * 16 + 8 * h + m] = myGv;
  __syncthreads();

  // ---- merge: out = fp*g0 + mean*(1-g0), full f32, coalesced ----
  const float g0v  = gbuf[wv * 16 + m];
  const float g1v  = 1.0f - g0v;
  const float* fpp = fp + fpBase;
  const float* mp  = mean + pix0 + m;
  float* op        = out + fpBase;
  #pragma unroll 4
  for (int c = h; c < 256; c += 2) {
    const long off = (long)c * HW;             // constant multiples -> imm offs
    const float r = fpp[off] * g0v + mp[off] * g1v;
    __builtin_nontemporal_store(r, op + off);  // write-only stream: TH=NT
  }
}

extern "C" void kernel_launch(void* const* d_in, const int* in_sizes, int n_in,
                              void* d_out, int out_size, void* d_ws, size_t ws_size,
                              hipStream_t stream) {
  (void)in_sizes; (void)n_in; (void)d_ws; (void)ws_size; (void)out_size;
  const float* fp3   = (const float*)d_in[0];
  const float* fp4   = (const float*)d_in[1];
  // d_in[2] = images_tensor: shape-checked only, unused in compute
  const float* mean3 = (const float*)d_in[3];
  const float* mean4 = (const float*)d_in[4];
  float* out = (float*)d_out;

  // ---- p3: HW = 96*160 = 15360, 960 tiles, 4 waves/block ----
  {
    constexpr int HW = 96 * 160;
    dim3 grid((8 * (HW / 16)) / 4);            // 1920 blocks
    gate_merge_kernel<HW><<<grid, 128, 0, stream>>>(
        fp3, mean3,
        (const float*)d_in[5],  (const float*)d_in[6],  (const float*)d_in[7],
        (const float*)d_in[8],  (const float*)d_in[9],  (const float*)d_in[10],
        (const float*)d_in[11], (const float*)d_in[12], (const float*)d_in[13],
        (const float*)d_in[14], (const float*)d_in[15], (const float*)d_in[16],
        out);
  }
  // ---- p4: HW = 48*80 = 3840, 240 tiles ----
  {
    constexpr int HW = 48 * 80;
    dim3 grid((8 * (HW / 16)) / 4);            // 480 blocks
    float* out4 = out + (size_t)8 * 256 * 96 * 160;
    gate_merge_kernel<HW><<<grid, 128, 0, stream>>>(
        fp4, mean4,
        (const float*)d_in[17], (const float*)d_in[18], (const float*)d_in[19],
        (const float*)d_in[20], (const float*)d_in[21], (const float*)d_in[22],
        (const float*)d_in[23], (const float*)d_in[24], (const float*)d_in[25],
        (const float*)d_in[26], (const float*)d_in[27], (const float*)d_in[28],
        out4);
  }
}
